// SRResNet_10_79800492359902
// MI455X (gfx1250) — matrix operand order repaired and checked by reading, not compiled, not run on hardware
//
#include <hip/hip_runtime.h>
#include <hip/hip_bf16.h>

typedef _Float16 v16h __attribute__((ext_vector_type(16)));
typedef _Float16 v8h  __attribute__((ext_vector_type(8)));
typedef float    v8f  __attribute__((ext_vector_type(8)));

#define NBLK    10
#define CH      64
#define CONDCH  128
#define BATCH   4
#define PW      66           // patch cols: 64-pixel tile + 2 halo

// ---------------------------------------------------------------------------
// CDNA5 async global->LDS copy (ASYNCcnt-tracked) + wait
// ---------------------------------------------------------------------------
__device__ __forceinline__ void async_load_b128(unsigned lds_off, const void* gptr) {
    asm volatile("global_load_async_to_lds_b128 %0, %1, off"
                 :: "v"(lds_off), "v"(gptr) : "memory");
}
__device__ __forceinline__ void wait_async0() {
    asm volatile("s_wait_asynccnt 0x0" ::: "memory");
}

// ---------------------------------------------------------------------------
// 20 EqualLinears: cond[m][b][o] = s_lin * dot(cf[b,:], mod_w[m,o,:]) + mod_b[m,o]
// ---------------------------------------------------------------------------
__global__ void cond_kernel(const float* __restrict__ cf, const float* __restrict__ mw,
                            const float* __restrict__ mb, float* __restrict__ out) {
    int idx = blockIdx.x * blockDim.x + threadIdx.x;              // [m][b][o]
    if (idx >= 2 * NBLK * BATCH * CH) return;
    int o = idx % CH;
    int b = (idx / CH) % BATCH;
    int m = idx / (CH * BATCH);
    const float s_lin = 0.08838834764831845f;                     // 1/sqrt(128)
    const float* wrow = mw + ((size_t)m * CH + o) * CONDCH;
    const float* crow = cf + (size_t)b * CONDCH;
    float acc = 0.f;
    for (int c = 0; c < CONDCH; ++c) acc += crow[c] * wrow[c];
    out[idx] = acc * s_lin + mb[m * CH + o];
}

// ---------------------------------------------------------------------------
// Shared-weight prep: f32 [rows][IC][3][3] -> f16 [rows][Kp], k = tap*IC+ic.
// ---------------------------------------------------------------------------
__global__ void prep_conv_w(const float* __restrict__ src, _Float16* __restrict__ dst,
                            int OC, int IC, int Kp, int rows, float scale) {
    int idx = blockIdx.x * blockDim.x + threadIdx.x;
    if (idx >= rows * Kp) return;
    int k = idx % Kp;
    int o = idx / Kp;
    float v = 0.f;
    if (o < OC && k < 9 * IC) {
        int tap = k / IC, ic = k % IC;
        v = src[((size_t)(o * IC + ic)) * 9 + tap] * scale;
    }
    dst[idx] = (_Float16)v;
}

// ---------------------------------------------------------------------------
// Modulated-weight prep: Wmod[conv][b][o][k] = blk_w[conv][o][ic][tap]*s64*cond[conv][b][ic]
// ---------------------------------------------------------------------------
__global__ void prep_mod_w(const float* __restrict__ bw, const float* __restrict__ condAll,
                           _Float16* __restrict__ dst) {
    int idx = blockIdx.x * blockDim.x + threadIdx.x;
    if (idx >= 2 * NBLK * BATCH * CH * 576) return;
    int k    = idx % 576;
    int o    = (idx / 576) % CH;
    int b    = (idx / (576 * CH)) % BATCH;
    int conv = idx / (576 * CH * BATCH);
    int tap = k >> 6, ic = k & 63;
    const float s64 = 0.041666666666666664f;                      // 1/sqrt(576)
    float v = bw[(((size_t)conv * CH + o) * CH + ic) * 9 + tap] * s64
            * condAll[(conv * BATCH + b) * CH + ic];
    dst[idx] = (_Float16)v;
}

// ---------------------------------------------------------------------------
// Implicit-GEMM 3x3 conv via V_WMMA_F32_16X16X32_F16, NHWC f16 activations.
// Workgroup: 8 wave32 = 64 out-channels x 64 pixels; each wave computes a
// 32x16 pair of C tiles (2 WMMAs share one B fragment per K-step).
//   grid = (W/64, H, b*ocGroups + ocg), ocGroups = 1<<ocgBits
// dstMode: 0 = f16 NHWC (+optional residual), 1 = f16 NHWC pixel-shuffle(2),
//          2 = f32 NCHW final output (oc < OC only)
// ---------------------------------------------------------------------------
template<typename TIn, int IC>
__global__ __launch_bounds__(256)
void conv_wmma(const TIn* __restrict__ src, int H, int W,
               const _Float16* __restrict__ Wg, int wBStride, int Kp, int wRows,
               const float* __restrict__ bias, int bias_mode,
               const _Float16* __restrict__ resid,
               void* dstv, int dstMode, int relu, int OC, int ocgBits) {
    __shared__ __align__(32) _Float16 patch[3 * PW * IC];

    const int wseg = blockIdx.x;
    const int h    = blockIdx.y;
    const int bz   = blockIdx.z;
    const int b    = bz >> ocgBits;
    const int ocg  = bz & ((1 << ocgBits) - 1);
    const int tid  = threadIdx.x;
    const int w0   = wseg * 64;

    if constexpr (IC == 64) {
        // ---- NHWC async-DMA patch fill: 3 contiguous PWx64-half row spans ----
        v8h z8 = {};
        v8h* pv = (v8h*)patch;                      // PW*8 = 528 v8h per patch row
        if (h == 0)       for (int i = tid; i < 528; i += 256) pv[i] = z8;
        if (h == H - 1)   for (int i = tid; i < 528; i += 256) pv[1056 + i] = z8;
        if (w0 == 0       && tid < 24) pv[(tid >> 3) * 528 + (tid & 7)] = z8;
        if (w0 + 64 >= W  && tid < 24) pv[(tid >> 3) * 528 + 65 * 8 + (tid & 7)] = z8;

        const int c0  = (w0 == 0) ? 1 : 0;          // first patch col from global
        const int c1  = (w0 + 64 < W) ? PW : PW - 1;
        const int nch = (c1 - c0) * 8;              // 16-B chunks per row
        for (int r = 0; r < 3; ++r) {
            int hi = h + r - 1;
            if (hi < 0 || hi >= H) continue;
            const _Float16* gbase = src + (((size_t)b * H + hi) * W + (w0 - 1 + c0)) * 64;
            unsigned lbase = (unsigned)(size_t)&patch[(r * PW + c0) * 64];
            for (int i = tid; i < nch; i += 256)
                async_load_b128(lbase + (unsigned)i * 16u, gbase + i * 8);
        }
        wait_async0();
    } else {
        // ---- small-IC fill from NCHW f32 input ----
        for (int i = tid; i < 3 * PW * IC; i += 256) {
            int ic = i % IC;
            int c  = (i / IC) % PW;
            int r  = i / (IC * PW);
            int hi = h + r - 1;
            int wi = w0 + c - 1;
            float v = 0.f;
            if (hi >= 0 && hi < H && wi >= 0 && wi < W)
                v = (float)src[(((size_t)b * IC + ic) * H + hi) * W + wi];
            patch[i] = (_Float16)v;
        }
    }
    __syncthreads();

    const int lane  = tid & 31;
    const int wave  = tid >> 5;
    const int mpair = wave & 1;                     // 2 M-pairs of 32 out-channels
    const int ntile = wave >> 1;                    // 4 N-tiles of 16 pixels
    const int hi16  = lane >> 4;
    const int rowA  = ocg * 64 + mpair * 32 + (lane & 15);   // first weight row
    const int koffA = hi16 * 16;                    // contiguous K offset (matches B)
    const int ncol  = ntile * 16 + (lane & 15);     // pixel within 64-seg

    const bool valid0 = rowA < wRows;               // wave-uniform (wRows mult of 16)
    const bool valid1 = rowA + 16 < wRows;
    const _Float16* wr0 = Wg + (size_t)b * wBStride + (size_t)rowA * Kp;
    const _Float16* wr1 = wr0 + (size_t)16 * Kp;
    __builtin_prefetch(wr0, 0, 1);                  // global_prefetch_b8
    const int Ktot = 9 * IC;

    v8f acc0 = {}, acc1 = {};
    #pragma unroll 2
    for (int k0 = 0; k0 < Kp; k0 += 32) {
        // B fragment (shared by both WMMAs): lanes 0-15 K=kb.., lanes 16-31 K=kb+16..
        v16h bfrag;
        const int kb = k0 + hi16 * 16;
        if constexpr (IC == 64) {
            int tap = kb >> 6;                      // 16-k run stays inside one tap
            int icb = kb & 63;
            bfrag = *(const v16h*)(&patch[((tap / 3) * PW + ncol + (tap % 3)) * 64 + icb]);
        } else {
            #pragma unroll
            for (int j = 0; j < 16; ++j) {
                int k = kb + j;
                _Float16 v = (_Float16)0.f;
                if (k < Ktot) {
                    int tap = k / IC, ic = k % IC;
                    v = patch[((tap / 3) * PW + ncol + (tap % 3)) * IC + ic];
                }
                bfrag[j] = v;
            }
        }
        // Two A fragments (two 16-row blocks), contiguous K order (k = 16*hi16 + s), same as B
        v16h a0 = {}, a1 = {};
        if (valid0) {
            a0 = *(const v16h*)(wr0 + k0 + koffA);
        }
        if (valid1) {
            a1 = *(const v16h*)(wr1 + k0 + koffA);
        }
        acc0 = __builtin_amdgcn_wmma_f32_16x16x32_f16(false, a0, false, bfrag,
                                                      (short)0, acc0, false, false);
        acc1 = __builtin_amdgcn_wmma_f32_16x16x32_f16(false, a1, false, bfrag,
                                                      (short)0, acc1, false, false);
    }

    // ---- store: C/D layout VGPR r -> M = r + 8*hi16, N = lane%16 ----
    const int col = w0 + ncol;
    auto store_tile = [&](const v8f& acc, int t) {
        const int ocBase = ocg * 64 + mpair * 32 + t * 16 + hi16 * 8;  // 8 consecutive oc
        if (dstMode == 0) {
            size_t base = (((size_t)b * H + h) * W + col) * (size_t)OC + ocBase;
            v8h rv = {};
            if (resid) rv = *(const v8h*)(resid + base);
            v8h ov;
            #pragma unroll
            for (int r = 0; r < 8; ++r) {
                float v = acc[r];
                if (bias) {
                    int m_local = ocBase - ocg * 64 + r;
                    v += (bias_mode == 1) ? bias[(b * 64 + m_local) >> 2]  // repeat_interleave
                                          : bias[ocBase + r];
                }
                if (relu) v = v > 0.f ? v : 0.f;
                if (resid) v += (float)rv[r];
                ov[r] = (_Float16)v;
            }
            *(v8h*)((_Float16*)dstv + base) = ov;            // single b128 store
        } else if (dstMode == 1) {
            const int H2 = H * 2, W2 = W * 2;
            #pragma unroll
            for (int r = 0; r < 8; ++r) {
                int oc = ocBase + r;
                float v = acc[r];
                if (bias) v += bias[oc];
                if (relu) v = v > 0.f ? v : 0.f;
                int C4 = oc >> 2, dr = (oc >> 1) & 1, dw = oc & 1;
                ((_Float16*)dstv)[(((size_t)b * H2 + (2 * h + dr)) * W2 + (2 * col + dw)) * 64
                                  + C4] = (_Float16)v;
            }
        } else {
            #pragma unroll
            for (int r = 0; r < 8; ++r) {
                int oc = ocBase + r;
                if (oc < OC) {
                    float v = acc[r];
                    if (bias) v += bias[oc];
                    ((float*)dstv)[(((size_t)b * OC + oc) * H + h) * W + col] = v;
                }
            }
        }
    };
    store_tile(acc0, 0);
    store_tile(acc1, 1);
}

// ---------------------------------------------------------------------------
extern "C" void kernel_launch(void* const* d_in, const int* in_sizes, int n_in,
                              void* d_out, int out_size, void* d_ws, size_t ws_size,
                              hipStream_t stream) {
    const float* x     = (const float*)d_in[0];
    const float* cf    = (const float*)d_in[1];
    const float* in_w  = (const float*)d_in[2];
    const float* in_b  = (const float*)d_in[3];
    const float* blk_w = (const float*)d_in[4];
    const float* blk_b = (const float*)d_in[5];
    const float* mod_w = (const float*)d_in[6];
    const float* mod_b = (const float*)d_in[7];
    const float* up_w  = (const float*)d_in[8];
    const float* up_b  = (const float*)d_in[9];
    const float* out_w = (const float*)d_in[10];
    const float* out_b = (const float*)d_in[11];

    char* ws = (char*)d_ws;
    size_t off = 0;
    auto carve = [&](size_t bytes) {
        size_t o = off;
        off = (off + bytes + 255) & ~(size_t)255;
        return o;
    };
    float*     condAll = (float*)    (ws + carve(2 * NBLK * BATCH * CH * sizeof(float)));
    _Float16*  Win     = (_Float16*) (ws + carve((size_t)64 * 32 * 2));
    _Float16*  Wmod    = (_Float16*) (ws + carve((size_t)2 * NBLK * BATCH * CH * 576 * 2));
    _Float16*  Wup     = (_Float16*) (ws + carve((size_t)512 * 576 * 2));
    _Float16*  Wout    = (_Float16*) (ws + carve((size_t)16 * 576 * 2));
    _Float16*  fea     = (_Float16*) (ws + carve((size_t)BATCH * CH * 128 * 128 * 2));
    _Float16*  tmp     = (_Float16*) (ws + carve((size_t)BATCH * CH * 128 * 128 * 2));
    _Float16*  buf1    = (_Float16*) (ws + carve((size_t)BATCH * CH * 256 * 256 * 2));
    _Float16*  buf2    = (_Float16*) (ws + carve((size_t)BATCH * CH * 512 * 512 * 2));

    const float s_in = 0.19245008972987526f;   // 1/sqrt(27)
    const float s64  = 0.041666666666666664f;  // 1/sqrt(576)

    cond_kernel<<<(2 * NBLK * BATCH * CH + 255) / 256, 256, 0, stream>>>(cf, mod_w, mod_b, condAll);

    prep_conv_w<<<(64 * 32 + 255) / 256, 256, 0, stream>>>(in_w, Win, 64, 3, 32, 64, s_in);
    prep_mod_w<<<(2 * NBLK * BATCH * CH * 576 + 255) / 256, 256, 0, stream>>>(blk_w, condAll, Wmod);
    prep_conv_w<<<(512 * 576 + 255) / 256, 256, 0, stream>>>(up_w, Wup, 512, 64, 576, 512, s64);
    prep_conv_w<<<(16 * 576 + 255) / 256, 256, 0, stream>>>(out_w, Wout, 3, 64, 576, 16, s64);

    // input conv 3->64 (NCHW f32 src -> NHWC f16), no activation
    conv_wmma<float, 3><<<dim3(2, 128, 4), 256, 0, stream>>>(
        x, 128, 128, Win, 0, 32, 64, in_b, 0, nullptr, fea, 0, 0, 64, 0);

    // 10 residual blocks of modulated convs (per-batch prescaled weights)
    const size_t wConv = (size_t)CH * 576;           // per-(conv,b) weight slab
    for (int i = 0; i < NBLK; ++i) {
        conv_wmma<_Float16, 64><<<dim3(2, 128, 4), 256, 0, stream>>>(
            fea, 128, 128, Wmod + (size_t)(2 * i) * BATCH * wConv, (int)wConv, 576, 64,
            blk_b + (2 * i) * 64, 1, nullptr, tmp, 0, 1, 64, 0);
        conv_wmma<_Float16, 64><<<dim3(2, 128, 4), 256, 0, stream>>>(
            tmp, 128, 128, Wmod + (size_t)(2 * i + 1) * BATCH * wConv, (int)wConv, 576, 64,
            blk_b + (2 * i + 1) * 64, 1, fea, fea, 0, 0, 64, 0);
    }

    // two fused conv + PixelShuffle(2) stages (NHWC)
    conv_wmma<_Float16, 64><<<dim3(2, 128, 16), 256, 0, stream>>>(
        fea, 128, 128, Wup, 0, 576, 256, up_b, 0, nullptr, buf1, 1, 0, 256, 2);
    conv_wmma<_Float16, 64><<<dim3(4, 256, 16), 256, 0, stream>>>(
        buf1, 256, 256, Wup + (size_t)256 * 576, 0, 576, 256, up_b + 256, 0, nullptr,
        buf2, 1, 0, 256, 2);

    // output conv 64->3, f32 NCHW store to d_out
    conv_wmma<_Float16, 64><<<dim3(8, 512, 4), 256, 0, stream>>>(
        buf2, 512, 512, Wout, 0, 576, 16, out_b, 0, nullptr, d_out, 2, 0, 3, 0);
}
